// BertGATRelationModel_24060406792244
// MI455X (gfx1250) — matrix operand order repaired and checked by reading, not compiled, not run on hardware
//
#include <hip/hip_runtime.h>
#include <hip/hip_bf16.h>
#include <math.h>

// ---------------- problem constants ----------------
#define B_    8
#define L_    256
#define H_    768
#define NL_   12
#define NH_   12
#define DH_   64
#define FF_   3072
#define GH_   4
#define GD_   128
#define GC_   (GH_*GD_)     // 512
#define NREL_ 41
#define CLS_IN_ 2816
#define M_    (B_*L_)       // 2048 rows of activations

typedef _Float16 v16h __attribute__((ext_vector_type(16)));
typedef _Float16 h8   __attribute__((ext_vector_type(8)));
typedef float    v8f  __attribute__((ext_vector_type(8)));

// ---------------- WMMA GEMM: C[M,N] = f16(A[M,K]) @ f16(B[K,N]) (+bias, +gelu) ----
// 256 threads = 8 wave32 waves arranged 4x2; block tile 128x128, wave tile 32x64,
// per k-step each wave issues 8 x v_wmma_f32_16x16x32_f16 from 2 A-frags x 4 B-frags.
// B is staged TRANSPOSED in LDS so per-lane fragment halves are K-contiguous
// (vectorizes to ds_load_b128, same as A). Staging is explicitly vectorized:
// A: 4x global b128 loads -> 2x ds_store_b128; B: 16 coalesced b32 loads issued
// as one clause -> 2x ds_store_b128 (no per-element load/wait serialization).
#define BM 128
#define BN 128
#define BK 32
#define LDA_S 40   // BK + 8 pad (halves); row stride 80 B (16B aligned)
#define LDB_S 40   // BK + 8 pad (halves), B stored [n][k]

__global__ __launch_bounds__(256)
void gemm_wmma_f16(const float* __restrict__ A, const float* __restrict__ Bm,
                   const float* __restrict__ bias, float* __restrict__ C,
                   int Mdim, int Ndim, int Kdim, int mode /*0=none,1=gelu*/)
{
  __shared__ _Float16 sA[BM * LDA_S];
  __shared__ _Float16 sBt[BN * LDB_S];   // transposed: [n][k]

  const int tid  = threadIdx.x;
  const int lane = tid & 31;
  const int wave = tid >> 5;       // 0..7
  const int wr   = wave >> 1;      // 0..3 -> 4 x 32 rows
  const int wc   = wave & 1;       // 0..1 -> 2 x 64 cols
  const int rowBase = blockIdx.y * BM;
  const int colBase = blockIdx.x * BN;

  // staging coordinates (fixed per thread)
  const int aRow = tid >> 1;            // 0..127
  const int aOff = (tid & 1) * 16;      // 0 or 16 (floats within the 32-wide k tile)
  const int bN   = tid & 127;           // 0..127
  const int bKo  = (tid >> 7) * 16;     // 0 or 16

  v8f acc[2][4] = {};

  for (int kb = 0; kb < Kdim; kb += BK) {
    if (kb + BK < Kdim) {   // global_prefetch_b8 of next k-tile
      __builtin_prefetch(&A[(size_t)(rowBase + aRow) * Kdim + kb + BK + aOff], 0, 1);
      __builtin_prefetch(&Bm[(size_t)(kb + BK + bKo) * Ndim + colBase + bN], 0, 1);
    }
    __syncthreads();

    // ---- stage A tile: 4x b128 global loads -> convert -> 2x b128 LDS stores ----
    {
      const float4* ga = (const float4*)&A[(size_t)(rowBase + aRow) * Kdim + kb + aOff];
      float4 f0 = ga[0], f1 = ga[1], f2 = ga[2], f3 = ga[3];
      h8 h0, h1;
      h0[0] = (_Float16)f0.x; h0[1] = (_Float16)f0.y; h0[2] = (_Float16)f0.z; h0[3] = (_Float16)f0.w;
      h0[4] = (_Float16)f1.x; h0[5] = (_Float16)f1.y; h0[6] = (_Float16)f1.z; h0[7] = (_Float16)f1.w;
      h1[0] = (_Float16)f2.x; h1[1] = (_Float16)f2.y; h1[2] = (_Float16)f2.z; h1[3] = (_Float16)f2.w;
      h1[4] = (_Float16)f3.x; h1[5] = (_Float16)f3.y; h1[6] = (_Float16)f3.z; h1[7] = (_Float16)f3.w;
      *(h8*)&sA[aRow * LDA_S + aOff]     = h0;
      *(h8*)&sA[aRow * LDA_S + aOff + 8] = h1;
    }
    // ---- stage B tile transposed: 16 coalesced b32 loads (one clause) ----
    {
      float tmp[16];
      #pragma unroll
      for (int kk = 0; kk < 16; ++kk)
        tmp[kk] = Bm[(size_t)(kb + bKo + kk) * Ndim + colBase + bN];
      h8 h0, h1;
      #pragma unroll
      for (int e = 0; e < 8; ++e) { h0[e] = (_Float16)tmp[e]; h1[e] = (_Float16)tmp[8 + e]; }
      *(h8*)&sBt[bN * LDB_S + bKo]     = h0;
      *(h8*)&sBt[bN * LDB_S + bKo + 8] = h1;
    }
    __syncthreads();

    // ---- fragment loads per CDNA5 16-bit WMMA VGPR layouts ----
    const int m   = lane & 15;
    const int ksA = (lane >> 4) << 3;   // lanes 16-31 hold K+8
    const int ksB = (lane >> 4) << 3;   // lanes 16-31 hold K+8
    v16h a[2], b[4];
    #pragma unroll
    for (int ti = 0; ti < 2; ++ti) {
      const _Float16* pa = &sA[(wr * 32 + ti * 16 + m) * LDA_S + ksA];
      #pragma unroll
      for (int e = 0; e < 16; ++e)
        a[ti][e] = pa[((e >> 3) << 4) + (e & 7)];     // e0..7 -> K+0..7, e8..15 -> K+16..23
    }
    #pragma unroll
    for (int tj = 0; tj < 4; ++tj) {
      const _Float16* pb = &sBt[(wc * 64 + tj * 16 + m) * LDB_S + ksB];
      #pragma unroll
      for (int e = 0; e < 16; ++e)
        b[tj][e] = pb[((e >> 3) << 4) + (e & 7)];     // e0..7 -> K+0..7, e8..15 -> K+16..23
    }
    #pragma unroll
    for (int ti = 0; ti < 2; ++ti)
      #pragma unroll
      for (int tj = 0; tj < 4; ++tj)
        acc[ti][tj] = __builtin_amdgcn_wmma_f32_16x16x32_f16(
            false, a[ti], false, b[tj], (short)0, acc[ti][tj], false, false);
  }

  // ---- epilogue: C/D layout M = vgpr + 8*(lane>=16), N = lane%16 ----
  const int mOff = (lane >> 4) * 8;
  const int nOff = lane & 15;
  #pragma unroll
  for (int ti = 0; ti < 2; ++ti) {
    #pragma unroll
    for (int tj = 0; tj < 4; ++tj) {
      int col  = colBase + wc * 64 + tj * 16 + nOff;
      int rowT = rowBase + wr * 32 + ti * 16 + mOff;
      #pragma unroll
      for (int r = 0; r < 8; ++r) {
        int rr = rowT + r;
        float vv = acc[ti][tj][r];
        if (bias) vv += bias[col];
        if (mode == 1) vv = 0.5f * vv * (1.0f + erff(vv * 0.70710678118f)); // exact gelu
        if (rr < Mdim && col < Ndim) C[(size_t)rr * Ndim + col] = vv;
      }
    }
  }
}

// ---------------- embeddings + LayerNorm ----------------
__global__ __launch_bounds__(256)
void embed_ln_kernel(const int* __restrict__ ids, const float* __restrict__ wemb,
                     const float* __restrict__ pemb, const float* __restrict__ temb,
                     const float* __restrict__ g, const float* __restrict__ bta,
                     float* __restrict__ x)
{
  int row = blockIdx.x, l = row & (L_ - 1), t = threadIdx.x;
  __shared__ float buf[H_];
  __shared__ float red[256];
  int id = ids[row];
  for (int j = t; j < H_; j += 256)
    buf[j] = wemb[(size_t)id * H_ + j] + pemb[(size_t)l * H_ + j] + temb[j];
  __syncthreads();
  float s = 0.f; for (int j = t; j < H_; j += 256) s += buf[j];
  red[t] = s; __syncthreads();
  for (int o = 128; o; o >>= 1) { if (t < o) red[t] += red[t + o]; __syncthreads(); }
  float mean = red[0] / H_; __syncthreads();
  float vs = 0.f; for (int j = t; j < H_; j += 256) { float d = buf[j] - mean; vs += d * d; }
  red[t] = vs; __syncthreads();
  for (int o = 128; o; o >>= 1) { if (t < o) red[t] += red[t + o]; __syncthreads(); }
  float inv = rsqrtf(red[0] / H_ + 1e-12f); __syncthreads();
  for (int j = t; j < H_; j += 256)
    x[(size_t)row * H_ + j] = (buf[j] - mean) * inv * g[j] + bta[j];
}

// ---------------- residual add + LayerNorm (in-place on x) ----------------
__global__ __launch_bounds__(256)
void add_ln_kernel(float* __restrict__ x, const float* __restrict__ tadd,
                   const float* __restrict__ g, const float* __restrict__ bta)
{
  int row = blockIdx.x, t = threadIdx.x;
  __shared__ float buf[H_];
  __shared__ float red[256];
  for (int j = t; j < H_; j += 256)
    buf[j] = x[(size_t)row * H_ + j] + tadd[(size_t)row * H_ + j];
  __syncthreads();
  float s = 0.f; for (int j = t; j < H_; j += 256) s += buf[j];
  red[t] = s; __syncthreads();
  for (int o = 128; o; o >>= 1) { if (t < o) red[t] += red[t + o]; __syncthreads(); }
  float mean = red[0] / H_; __syncthreads();
  float vs = 0.f; for (int j = t; j < H_; j += 256) { float d = buf[j] - mean; vs += d * d; }
  red[t] = vs; __syncthreads();
  for (int o = 128; o; o >>= 1) { if (t < o) red[t] += red[t + o]; __syncthreads(); }
  float inv = rsqrtf(red[0] / H_ + 1e-12f); __syncthreads();
  for (int j = t; j < H_; j += 256)
    x[(size_t)row * H_ + j] = (buf[j] - mean) * inv * g[j] + bta[j];
}

// ---------------- attention: one block per (b,h,q), softmax over 256 keys -----
__global__ __launch_bounds__(256)
void attention_kernel(const float* __restrict__ q, const float* __restrict__ k,
                      const float* __restrict__ v, const int* __restrict__ mask,
                      float* __restrict__ o)
{
  int bx = blockIdx.x;
  int qi = bx & (L_ - 1);
  int h  = (bx >> 8) % NH_;
  int b  = bx / (L_ * NH_);
  int t  = threadIdx.x;
  __shared__ float sq[DH_];
  __shared__ float sp[L_];
  __shared__ float red[256];
  size_t qoff = (size_t)(b * L_ + qi) * H_ + h * DH_;
  if (t < DH_) sq[t] = q[qoff + t];
  __syncthreads();
  const float* kr = k + (size_t)(b * L_ + t) * H_ + h * DH_;
  float s = 0.f;
  #pragma unroll 8
  for (int d = 0; d < DH_; ++d) s += sq[d] * kr[d];
  s *= 0.125f;                                      // 1/sqrt(64)
  s += (1.f - (float)mask[b * L_ + t]) * -10000.f;
  red[t] = s; __syncthreads();
  for (int o2 = 128; o2; o2 >>= 1) { if (t < o2) red[t] = fmaxf(red[t], red[t + o2]); __syncthreads(); }
  float mx = red[0]; __syncthreads();
  float p = expf(s - mx); sp[t] = p;
  red[t] = p; __syncthreads();
  for (int o2 = 128; o2; o2 >>= 1) { if (t < o2) red[t] += red[t + o2]; __syncthreads(); }
  float dn = red[0]; __syncthreads();
  if (t < DH_) {
    float acc = 0.f;
    for (int kk = 0; kk < L_; ++kk)
      acc += sp[kk] * v[(size_t)(b * L_ + kk) * H_ + h * DH_ + t];
    o[qoff + t] = acc / dn;
  }
}

// ---------------- pooler: tanh(x[:,0] @ pool_w + pool_b) ----------------
__global__ __launch_bounds__(256)
void pooler_kernel(const float* __restrict__ x, const float* __restrict__ pw,
                   const float* __restrict__ pb, float* __restrict__ pooled)
{
  int b = blockIdx.x, t = threadIdx.x;
  __shared__ float sx[H_];
  for (int j = t; j < H_; j += 256) sx[j] = x[(size_t)(b * L_) * H_ + j];
  __syncthreads();
  for (int c = t; c < H_; c += 256) {
    float acc = pb[c];
    for (int j = 0; j < H_; ++j) acc += sx[j] * pw[(size_t)j * H_ + c];
    pooled[b * H_ + c] = tanhf(acc);
  }
}

// ---------------- GAT helpers ----------------
__device__ __forceinline__ unsigned fenc(float f) {
  unsigned u = __float_as_uint(f);
  return (u & 0x80000000u) ? ~u : (u | 0x80000000u);
}
__device__ __forceinline__ float fdec(unsigned kk) {
  return (kk & 0x80000000u) ? __uint_as_float(kk ^ 0x80000000u) : __uint_as_float(~kk);
}

__global__ __launch_bounds__(128)
void gat_attn_coef(const float* __restrict__ xp, const float* __restrict__ att_src,
                   const float* __restrict__ att_dst, float* __restrict__ a_src,
                   float* __restrict__ a_dst)
{
  int n = blockIdx.x, t = threadIdx.x;
  __shared__ float red[128];
  for (int h = 0; h < GH_; ++h) {
    float vv = xp[(size_t)n * GC_ + h * GD_ + t];
    red[t] = vv * att_src[h * GD_ + t]; __syncthreads();
    for (int o = 64; o; o >>= 1) { if (t < o) red[t] += red[t + o]; __syncthreads(); }
    if (t == 0) a_src[n * GH_ + h] = red[0];
    __syncthreads();
    red[t] = vv * att_dst[h * GD_ + t]; __syncthreads();
    for (int o = 64; o; o >>= 1) { if (t < o) red[t] += red[t + o]; __syncthreads(); }
    if (t == 0) a_dst[n * GH_ + h] = red[0];
    __syncthreads();
  }
}

__global__ __launch_bounds__(256)
void gat_init(float* __restrict__ agg, float* __restrict__ den,
              unsigned* __restrict__ mkey, int nAgg, int nHN)
{
  int i = blockIdx.x * 256 + threadIdx.x;
  if (i < nAgg) agg[i] = 0.f;
  if (i < nHN) { den[i] = 0.f; mkey[i] = fenc(-3.0e38f); }
}

__global__ __launch_bounds__(256)
void gat_edge_max(const int* __restrict__ ei, int E, int Etot,
                  const float* __restrict__ a_src, const float* __restrict__ a_dst,
                  unsigned* __restrict__ mkey)
{
  int i = blockIdx.x * 256 + threadIdx.x;
  if (i >= Etot) return;
  int s, d;
  if (i < E) { s = ei[i]; d = ei[E + i]; } else { s = d = i - E; }
  #pragma unroll
  for (int h = 0; h < GH_; ++h) {
    float e = a_src[s * GH_ + h] + a_dst[d * GH_ + h];
    e = e > 0.f ? e : 0.2f * e;   // leaky_relu(0.2)
    atomicMax(&mkey[d * GH_ + h], fenc(e));
  }
}

__global__ __launch_bounds__(256)
void gat_edge_sum(const int* __restrict__ ei, int E, int Etot,
                  const float* __restrict__ a_src, const float* __restrict__ a_dst,
                  const unsigned* __restrict__ mkey, float* __restrict__ den)
{
  int i = blockIdx.x * 256 + threadIdx.x;
  if (i >= Etot) return;
  int s, d;
  if (i < E) { s = ei[i]; d = ei[E + i]; } else { s = d = i - E; }
  #pragma unroll
  for (int h = 0; h < GH_; ++h) {
    float e = a_src[s * GH_ + h] + a_dst[d * GH_ + h];
    e = e > 0.f ? e : 0.2f * e;
    atomicAdd(&den[d * GH_ + h], expf(e - fdec(mkey[d * GH_ + h])));
  }
}

__global__ __launch_bounds__(128)
void gat_edge_agg(const int* __restrict__ ei, int E, int Etot,
                  const float* __restrict__ a_src, const float* __restrict__ a_dst,
                  const unsigned* __restrict__ mkey, const float* __restrict__ den,
                  const float* __restrict__ xp, float* __restrict__ agg)
{
  int i = blockIdx.x;
  if (i >= Etot) return;
  int s, d;
  if (i < E) { s = ei[i]; d = ei[E + i]; } else { s = d = i - E; }
  int t = threadIdx.x;
  #pragma unroll
  for (int h = 0; h < GH_; ++h) {
    float e = a_src[s * GH_ + h] + a_dst[d * GH_ + h];
    e = e > 0.f ? e : 0.2f * e;
    float alpha = expf(e - fdec(mkey[d * GH_ + h])) / den[d * GH_ + h];
    atomicAdd(&agg[(size_t)d * GC_ + h * GD_ + t],
              alpha * xp[(size_t)s * GC_ + h * GD_ + t]);
  }
}

__global__ __launch_bounds__(256)
void gat_mean_kernel(const float* __restrict__ agg, const float* __restrict__ gbias,
                     float* __restrict__ gmean)
{
  int b = blockIdx.x;
  int c = blockIdx.y * 256 + threadIdx.x;   // 512 features -> gridDim.y == 2
  float s = 0.f;
  for (int l = 0; l < L_; ++l) s += agg[(size_t)(b * L_ + l) * GC_ + c];
  gmean[b * GC_ + c] = s / (float)L_ + gbias[c];
}

// ---------------- classifier head ----------------
__global__ __launch_bounds__(256)
void classifier_kernel(const float* __restrict__ x, const int* __restrict__ subj,
                       const int* __restrict__ obj, const float* __restrict__ pooled,
                       const float* __restrict__ gmean, const float* __restrict__ w1,
                       const float* __restrict__ b1, const float* __restrict__ w2,
                       const float* __restrict__ b2, float* __restrict__ out)
{
  int b = blockIdx.x, t = threadIdx.x;
  __shared__ float cat[CLS_IN_];
  __shared__ float h1[256];
  int sp = subj[b], op = obj[b];
  for (int j = t; j < H_; j += 256) {
    cat[j]          = x[(size_t)(b * L_ + sp) * H_ + j];
    cat[H_ + j]     = x[(size_t)(b * L_ + op) * H_ + j];
    cat[2 * H_ + j] = pooled[b * H_ + j];
  }
  for (int j = t; j < GC_; j += 256) cat[3 * H_ + j] = gmean[b * GC_ + j];
  __syncthreads();
  float acc = b1[t];
  for (int j = 0; j < CLS_IN_; ++j) acc += cat[j] * w1[(size_t)j * 256 + t];
  h1[t] = fmaxf(acc, 0.f);
  __syncthreads();
  if (t < NREL_) {
    float a2 = b2[t];
    for (int j = 0; j < 256; ++j) a2 += h1[j] * w2[(size_t)j * NREL_ + t];
    out[b * NREL_ + t] = a2;
  }
}

// ---------------- host-side orchestration ----------------
extern "C" void kernel_launch(void* const* d_in, const int* in_sizes, int n_in,
                              void* d_out, int out_size, void* d_ws, size_t ws_size,
                              hipStream_t stream)
{
  (void)n_in; (void)out_size; (void)ws_size;
  const int* ids   = (const int*)d_in[0];
  const int* amask = (const int*)d_in[1];
  const int* subj  = (const int*)d_in[2];
  const int* obj   = (const int*)d_in[3];
  const int* eidx  = (const int*)d_in[4];
  const int  E     = in_sizes[4] / 2;

  const float* wemb = (const float*)d_in[5];
  const float* pemb = (const float*)d_in[6];
  const float* temb = (const float*)d_in[7];
  const float* elg  = (const float*)d_in[8];
  const float* elb  = (const float*)d_in[9];
  const float* wq = (const float*)d_in[10]; const float* bq = (const float*)d_in[11];
  const float* wk = (const float*)d_in[12]; const float* bk = (const float*)d_in[13];
  const float* wv = (const float*)d_in[14]; const float* bv = (const float*)d_in[15];
  const float* wo = (const float*)d_in[16]; const float* bo = (const float*)d_in[17];
  const float* l1g = (const float*)d_in[18]; const float* l1b = (const float*)d_in[19];
  const float* w1 = (const float*)d_in[20]; const float* b1 = (const float*)d_in[21];
  const float* w2 = (const float*)d_in[22]; const float* b2 = (const float*)d_in[23];
  const float* l2g = (const float*)d_in[24]; const float* l2b = (const float*)d_in[25];
  const float* poolw = (const float*)d_in[26]; const float* poolb = (const float*)d_in[27];
  const float* gatw  = (const float*)d_in[28];
  const float* attsrc = (const float*)d_in[29]; const float* attdst = (const float*)d_in[30];
  const float* gbias  = (const float*)d_in[31];
  const float* cw1 = (const float*)d_in[32]; const float* cb1 = (const float*)d_in[33];
  const float* cw2 = (const float*)d_in[34]; const float* cb2 = (const float*)d_in[35];

  // workspace carve (floats)
  float* ws = (float*)d_ws;
  size_t off = 0;
  float* x   = ws + off; off += (size_t)M_ * H_;
  float* qb  = ws + off; off += (size_t)M_ * H_;
  float* kb2 = ws + off; off += (size_t)M_ * H_;
  float* vb  = ws + off; off += (size_t)M_ * H_;
  float* t0  = ws + off; off += (size_t)M_ * H_;    // attn concat / mlp down out
  float* t1  = ws + off; off += (size_t)M_ * H_;    // attn proj out
  float* mid = ws + off; off += (size_t)M_ * FF_;
  float* xp  = ws + off; off += (size_t)M_ * GC_;
  float* agg = ws + off; off += (size_t)M_ * GC_;
  float* asrc = ws + off; off += (size_t)M_ * GH_;
  float* adst = ws + off; off += (size_t)M_ * GH_;
  float* den  = ws + off; off += (size_t)M_ * GH_;
  unsigned* mkey = (unsigned*)(ws + off); off += (size_t)M_ * GH_;
  float* pooled  = ws + off; off += (size_t)B_ * H_;
  float* gmean   = ws + off; off += (size_t)B_ * GC_;

  dim3 blk(256);
  embed_ln_kernel<<<M_, blk, 0, stream>>>(ids, wemb, pemb, temb, elg, elb, x);

  dim3 gH(H_ / BN, M_ / BM);    // 6 x 16
  dim3 gF(FF_ / BN, M_ / BM);   // 24 x 16
  for (int i = 0; i < NL_; ++i) {
    const float* Wq = wq + (size_t)i * H_ * H_;  const float* Bq = bq + (size_t)i * H_;
    const float* Wk = wk + (size_t)i * H_ * H_;  const float* Bk = bk + (size_t)i * H_;
    const float* Wv = wv + (size_t)i * H_ * H_;  const float* Bv = bv + (size_t)i * H_;
    const float* Wo = wo + (size_t)i * H_ * H_;  const float* Bo = bo + (size_t)i * H_;
    const float* W1 = w1 + (size_t)i * H_ * FF_; const float* B1 = b1 + (size_t)i * FF_;
    const float* W2 = w2 + (size_t)i * FF_ * H_; const float* B2 = b2 + (size_t)i * H_;

    gemm_wmma_f16<<<gH, blk, 0, stream>>>(x, Wq, Bq, qb,  M_, H_, H_, 0);
    gemm_wmma_f16<<<gH, blk, 0, stream>>>(x, Wk, Bk, kb2, M_, H_, H_, 0);
    gemm_wmma_f16<<<gH, blk, 0, stream>>>(x, Wv, Bv, vb,  M_, H_, H_, 0);
    attention_kernel<<<B_ * NH_ * L_, blk, 0, stream>>>(qb, kb2, vb, amask, t0);
    gemm_wmma_f16<<<gH, blk, 0, stream>>>(t0, Wo, Bo, t1, M_, H_, H_, 0);
    add_ln_kernel<<<M_, blk, 0, stream>>>(x, t1, l1g + (size_t)i * H_, l1b + (size_t)i * H_);
    gemm_wmma_f16<<<gF, blk, 0, stream>>>(x, W1, B1, mid, M_, FF_, H_, 1);   // fused gelu
    gemm_wmma_f16<<<gH, blk, 0, stream>>>(mid, W2, B2, t0, M_, H_, FF_, 0);
    add_ln_kernel<<<M_, blk, 0, stream>>>(x, t0, l2g + (size_t)i * H_, l2b + (size_t)i * H_);
  }

  pooler_kernel<<<B_, blk, 0, stream>>>(x, poolw, poolb, pooled);

  // GAT
  dim3 gG(GC_ / BN, M_ / BM);   // 4 x 16
  gemm_wmma_f16<<<gG, blk, 0, stream>>>(x, gatw, nullptr, xp, M_, GC_, H_, 0);
  gat_attn_coef<<<M_, dim3(128), 0, stream>>>(xp, attsrc, attdst, asrc, adst);
  const int nAgg = M_ * GC_, nHN = M_ * GH_;
  gat_init<<<(nAgg + 255) / 256, blk, 0, stream>>>(agg, den, mkey, nAgg, nHN);
  const int Etot = E + M_;
  gat_edge_max<<<(Etot + 255) / 256, blk, 0, stream>>>(eidx, E, Etot, asrc, adst, mkey);
  gat_edge_sum<<<(Etot + 255) / 256, blk, 0, stream>>>(eidx, E, Etot, asrc, adst, mkey, den);
  gat_edge_agg<<<Etot, dim3(128), 0, stream>>>(eidx, E, Etot, asrc, adst, mkey, den, xp, agg);
  gat_mean_kernel<<<dim3(B_, 2), blk, 0, stream>>>(agg, gbias, gmean);

  classifier_kernel<<<B_, blk, 0, stream>>>(x, subj, obj, pooled, gmean,
                                            cw1, cb1, cw2, cb2, (float*)d_out);
}
